// PCPolicy_55448027791908
// MI455X (gfx1250) — matrix operand order repaired and checked by reading, not compiled, not run on hardware
//
#include <hip/hip_runtime.h>
#include <hip/hip_bf16.h>

// ---------------- problem constants ----------------
constexpr int Bsz    = 4;
constexpr int Nn     = 512;
constexpr int Esz    = 128;
constexpr int NODE_F = 16;
constexpr int EDGE_F = 8;
constexpr int TOT_F  = 22;   // NODE_F + 6 dyn
constexpr int Lnum   = 3;

typedef __attribute__((ext_vector_type(16))) __bf16 v16bf;
typedef __attribute__((ext_vector_type(8)))  float  v8f;

// tanh via hardware exp2 + rcp (VALU trans pipe); exact saturation at the ends.
static __device__ __forceinline__ float fast_tanh(float x) {
    x = fminf(fmaxf(x, -10.0f), 10.0f);
    float e = __builtin_amdgcn_exp2f(x * 2.88539008177793f);   // exp(2x)
    return (e - 1.0f) * __builtin_amdgcn_rcpf(e + 1.0f);
}

// ---------------- x = concat(node_features, dyn) ----------------
__global__ void build_x_kernel(const float* __restrict__ nf,
                               const int* __restrict__ assigned,
                               const int* __restrict__ open_group,
                               const int* __restrict__ action_mask,
                               const float* __restrict__ build_limit,
                               const float* __restrict__ open_group_size,
                               float* __restrict__ x) {
    int idx = blockIdx.x * blockDim.x + threadIdx.x;
    int total = Bsz * Nn * TOT_F;
    if (idx >= total) return;
    int f  = idx % TOT_F;
    int bn = idx / TOT_F;
    int b  = bn / Nn;
    float v;
    if (f < NODE_F)      v = nf[bn * NODE_F + f];
    else if (f == 16)    v = assigned[bn]    ? 1.0f : 0.0f;
    else if (f == 17)    v = open_group[bn]  ? 1.0f : 0.0f;
    else if (f == 18)    v = action_mask[bn] ? 1.0f : 0.0f;
    else {
        int k = f - 19;
        v = open_group_size[b * 3 + k] / fmaxf(build_limit[b * 3 + k], 1e-6f);
    }
    x[idx] = v;
}

// ---------------- pre-pack: fp32 MxK -> bf16 MxKp (zero-padded) ----------------
__global__ void pack_a_bf16(const float* __restrict__ A, __bf16* __restrict__ out,
                            int M, int K, int Kp) {
    int idx = blockIdx.x * blockDim.x + threadIdx.x;
    if (idx >= M * Kp) return;
    int m = idx / Kp, k = idx % Kp;
    out[idx] = (k < K) ? (__bf16)A[m * K + k] : (__bf16)0.0f;
}

// ---------------- pre-pack: fp32 KxN -> bf16 NxKp transposed (zero-padded) ----------------
__global__ void pack_bt_bf16(const float* __restrict__ Bw, __bf16* __restrict__ out,
                             int N, int K, int Kp) {
    int idx = blockIdx.x * blockDim.x + threadIdx.x;
    if (idx >= N * Kp) return;
    int n = idx / Kp, k = idx % Kp;
    out[idx] = (k < K) ? (__bf16)Bw[k * N + n] : (__bf16)0.0f;
}

// ---------------- bf16-WMMA GEMM: D = A(MxKp bf16) * Bt(NxKp bf16)^T [+bias][+Cin] ----------------
// One wave per 16x16 tile. All fragment loads are contiguous 128-bit loads:
//   A 16x32 bf16 : lane m = lane&15; runs K = k0+aHalf..+7 and k0+aHalf+16..+23
//   B 32x16 bf16 : lane n = lane&15; runs K = k0+bHalf..+7 and k0+bHalf+16..+23 (Bt is N-major)
//   C/D 16x16 f32: lane n = lane&15; vgpr r -> M = r + ((lane&16)?8:0)
template <int Kp, bool HasCin, bool HasBias>
__global__ void wmma_gemm_bf16_kernel(const __bf16* __restrict__ Abf,
                                      const __bf16* __restrict__ Bt,
                                      const float* __restrict__ Cin,
                                      const float* __restrict__ bias,
                                      float* __restrict__ D,
                                      int M, int N) {
    const int lane   = threadIdx.x & 31;
    const int wave   = threadIdx.x >> 5;
    const int tilesN = N >> 4;
    const int nTiles = (M >> 4) * tilesN;
    const int tile   = blockIdx.x * (blockDim.x >> 5) + wave;   // wave-uniform
    if (tile >= nTiles) return;                                 // uniform exit: EXEC all-1s

    const int tm = tile / tilesN;
    const int tn = tile % tilesN;

    const int row_a = tm * 16 + (lane & 15);
    const int col_b = tn * 16 + (lane & 15);
    const int aOff  = (lane & 16) ? 1 : 0;   // +8 bf16  in uint4 units
    const int bOff  = (lane & 16) ? 1 : 0;   // +8 bf16  in uint4 units

    const uint4* __restrict__ arow = (const uint4*)(Abf + (size_t)row_a * Kp);
    const uint4* __restrict__ brow = (const uint4*)(Bt  + (size_t)col_b * Kp);

    v8f acc = {};
#pragma unroll
    for (int k0 = 0; k0 < Kp; k0 += 32) {
        const int u0 = k0 >> 3;   // uint4 index (8 bf16 per uint4)
        union { v16bf v; uint4 u[2]; } a, b;
        a.u[0] = arow[u0 + aOff];        // K = k0+aHalf   .. +7   (t=0..7)
        a.u[1] = arow[u0 + aOff + 2];    // K = k0+aHalf+16.. +23  (t=8..15)
        b.u[0] = brow[u0 + bOff];        // K = k0+bHalf   .. +7
        b.u[1] = brow[u0 + bOff + 2];    // K = k0+bHalf+16.. +23
        acc = __builtin_amdgcn_wmma_f32_16x16x32_bf16(
            /*neg_a=*/false, a.v, /*neg_b=*/false, b.v,
            /*c_mod=*/(short)0, acc, /*reuse_a=*/false, /*reuse_b=*/false);
    }

    float bval = 0.0f;
    if constexpr (HasBias) bval = bias[col_b];

    const int mBase = tm * 16 + ((lane & 16) ? 8 : 0);
    size_t off = (size_t)mBase * N + col_b;
    float*       __restrict__ dptr = D + off;
    const float* __restrict__ cptr = HasCin ? (Cin + off) : nullptr;
#pragma unroll
    for (int r = 0; r < 8; ++r) {
        float v = acc[r] + bval;
        if constexpr (HasCin) v += cptr[0];
        dptr[0] = v;
        dptr += N;
        if constexpr (HasCin) cptr += N;
    }
}

// ---------------- edge message aggregation (VALU-dominant part) ----------------
// agg[b,i,e] = mean_j tanh( hi[b,i,e] + sum_f edge[b,i,j,f]*ew[f,e] + W[b,i,j]*mpw[e] )
__global__ void edge_agg_kernel(const float* __restrict__ hi,
                                const float* __restrict__ edge,
                                const float* __restrict__ W,
                                const float* __restrict__ ew_l,    // [8,128]
                                const float* __restrict__ mpw_l,   // [128]
                                float* __restrict__ agg) {
    constexpr int CH = 64;
    __shared__ float4 s_edge4[CH * EDGE_F / 4];   // 64 j's * 8 f = 512 floats
    __shared__ float  s_w[CH];
    float* s_edge = (float*)s_edge4;

    const int bi = blockIdx.x;            // b*N + i
    const int e  = threadIdx.x;           // 0..127
    const float hie = hi[bi * Esz + e];
    const float mpw = mpw_l[e];
    float ew[EDGE_F];
#pragma unroll
    for (int f = 0; f < EDGE_F; ++f) ew[f] = ew_l[f * Esz + e];

    const float* erow = edge + (size_t)bi * Nn * EDGE_F;
    const float* wrow = W    + (size_t)bi * Nn;

    float acc = 0.0f;
    for (int j0 = 0; j0 < Nn; j0 += CH) {
        __syncthreads();
        s_edge4[e] = ((const float4*)(erow + j0 * EDGE_F))[e];
        if (e < CH) s_w[e] = wrow[j0 + e];
        __syncthreads();
#pragma unroll 4
        for (int jj = 0; jj < CH; ++jj) {
            const float* ef = s_edge + jj * EDGE_F;
            float v = fmaf(s_w[jj], mpw, hie);
#pragma unroll
            for (int f = 0; f < EDGE_F; ++f) v = fmaf(ef[f], ew[f], v);
            acc += fast_tanh(v);
        }
    }
    agg[bi * Esz + e] = acc * (1.0f / (float)Nn);
}

// ---------------- attention head: context, query, logits ----------------
__global__ void attention_kernel(const float* __restrict__ h,
                                 const float* __restrict__ keys,
                                 const int* __restrict__ open_group,
                                 const int* __restrict__ action_mask,
                                 const float* __restrict__ context_w,
                                 const float* __restrict__ logit_bias,
                                 float* __restrict__ out) {
    __shared__ float s_ctx[Esz];
    __shared__ float s_q[Esz];
    const int b = blockIdx.x;
    const int e = threadIdx.x;

    float sum = 0.0f;
    int   cnt = 0;
    for (int n = 0; n < Nn; ++n) {
        int og = open_group[b * Nn + n];
        cnt += (og != 0);
        if (og) sum += h[((size_t)b * Nn + n) * Esz + e];
    }
    float ctx = (cnt > 0) ? (sum / fmaxf((float)cnt, 1.0f))
                          : h[((size_t)b * Nn + 0) * Esz + e];
    s_ctx[e] = ctx;
    __syncthreads();

    float q = 0.0f;
    for (int k = 0; k < Esz; ++k) q = fmaf(s_ctx[k], context_w[k * Esz + e], q);
    s_q[e] = q;
    __syncthreads();

    const float inv_sqrt_e = 0.08838834764831845f;  // 1/sqrt(128)
    const float lb = logit_bias[0];
    for (int n = e; n < Nn; n += Esz) {
        const float* kr = keys + ((size_t)b * Nn + n) * Esz;
        float dot = 0.0f;
        for (int k = 0; k < Esz; ++k) dot = fmaf(s_q[k], kr[k], dot);
        float lg = dot * inv_sqrt_e + lb;
        out[b * Nn + n] = action_mask[b * Nn + n] ? lg : -1e9f;
    }
}

// ---------------- host launcher ----------------
extern "C" void kernel_launch(void* const* d_in, const int* in_sizes, int n_in,
                              void* d_out, int out_size, void* d_ws, size_t ws_size,
                              hipStream_t stream) {
    const float* node_features   = (const float*)d_in[0];
    const float* edge_features   = (const float*)d_in[1];
    const float* W               = (const float*)d_in[2];
    const int*   assigned        = (const int*)d_in[3];
    const int*   open_group      = (const int*)d_in[4];
    const int*   action_mask     = (const int*)d_in[5];
    const float* build_limit     = (const float*)d_in[6];
    const float* open_group_size = (const float*)d_in[7];
    const float* node_embed_w    = (const float*)d_in[8];
    const float* node_embed_b    = (const float*)d_in[9];
    const float* mp_node_w       = (const float*)d_in[10];
    const float* mp_edge_w       = (const float*)d_in[11];
    const float* mp_weight_w     = (const float*)d_in[12];
    const float* mp_out_w        = (const float*)d_in[13];
    const float* mp_out_b        = (const float*)d_in[14];
    const float* context_w       = (const float*)d_in[15];
    const float* key_w           = (const float*)d_in[16];
    const float* logit_bias      = (const float*)d_in[17];
    float* out = (float*)d_out;

    const int BN = Bsz * Nn;                   // 2048

    // workspace layout (fp32 region then bf16 region; all 16B aligned)
    float* ws   = (float*)d_ws;
    float* x    = ws;                          // 45056
    float* h    = x    + BN * TOT_F;           // 262144
    float* hi   = h    + BN * Esz;             // 262144
    float* agg  = hi   + BN * Esz;             // 262144
    float* keys = agg  + BN * Esz;             // 262144
    __bf16* Abf = (__bf16*)(keys + BN * Esz);  // up to 2048*128 bf16 = 512KB
    __bf16* Btb = Abf + (size_t)BN * Esz;      // up to 128*128 bf16

    // 1) build x
    {
        int total = BN * TOT_F;
        build_x_kernel<<<(total + 255) / 256, 256, 0, stream>>>(
            node_features, assigned, open_group, action_mask,
            build_limit, open_group_size, x);
    }

    const int gemmBlocks = (BN / 16) * (Esz / 16) / 8;   // 1024 tiles / 8 waves = 128

    // 2) embed: h = x @ node_embed_w + b   (K=22 padded to 32)
    {
        int na = BN * 32, nb = Esz * 32;
        pack_a_bf16 <<<(na + 255) / 256, 256, 0, stream>>>(x, Abf, BN, TOT_F, 32);
        pack_bt_bf16<<<(nb + 255) / 256, 256, 0, stream>>>(node_embed_w, Btb, Esz, TOT_F, 32);
        wmma_gemm_bf16_kernel<32, false, true><<<gemmBlocks, 256, 0, stream>>>(
            Abf, Btb, nullptr, node_embed_b, h, BN, Esz);
    }

    // 3) message-passing layers
    for (int l = 0; l < Lnum; ++l) {
        const float* nw  = mp_node_w   + (size_t)l * Esz * Esz;
        const float* ew  = mp_edge_w   + (size_t)l * EDGE_F * Esz;
        const float* mpw = mp_weight_w + (size_t)l * Esz;
        const float* ow  = mp_out_w    + (size_t)l * Esz * Esz;
        const float* ob  = mp_out_b    + (size_t)l * Esz;
        int na = BN * Esz, nb = Esz * Esz;

        // hi = h @ mp_node_w[l]
        pack_a_bf16 <<<(na + 255) / 256, 256, 0, stream>>>(h, Abf, BN, Esz, Esz);
        pack_bt_bf16<<<(nb + 255) / 256, 256, 0, stream>>>(nw, Btb, Esz, Esz, Esz);
        wmma_gemm_bf16_kernel<Esz, false, false><<<gemmBlocks, 256, 0, stream>>>(
            Abf, Btb, nullptr, nullptr, hi, BN, Esz);

        // agg = mean_j tanh(hi + ej + wij)
        edge_agg_kernel<<<BN, Esz, 0, stream>>>(hi, edge_features, W, ew, mpw, agg);

        // h = h + agg @ mp_out_w[l] + mp_out_b[l]
        pack_a_bf16 <<<(na + 255) / 256, 256, 0, stream>>>(agg, Abf, BN, Esz, Esz);
        pack_bt_bf16<<<(nb + 255) / 256, 256, 0, stream>>>(ow, Btb, Esz, Esz, Esz);
        wmma_gemm_bf16_kernel<Esz, true, true><<<gemmBlocks, 256, 0, stream>>>(
            Abf, Btb, h, ob, h, BN, Esz);
    }

    // 4) keys = h @ key_w
    {
        int na = BN * Esz, nb = Esz * Esz;
        pack_a_bf16 <<<(na + 255) / 256, 256, 0, stream>>>(h, Abf, BN, Esz, Esz);
        pack_bt_bf16<<<(nb + 255) / 256, 256, 0, stream>>>(key_w, Btb, Esz, Esz, Esz);
        wmma_gemm_bf16_kernel<Esz, false, false><<<gemmBlocks, 256, 0, stream>>>(
            Abf, Btb, nullptr, nullptr, keys, BN, Esz);
    }

    // 5) attention + logits
    attention_kernel<<<Bsz, Esz, 0, stream>>>(
        h, keys, open_group, action_mask, context_w, logit_bias, out);
}
